// GPNN_VCOCO_4922032522013
// MI455X (gfx1250) — matrix operand order repaired and checked by reading, not compiled, not run on hardware
//
#include <hip/hip_runtime.h>
#include <hip/hip_bf16.h>
#include <cstdint>
#include <cstddef>

// CDNA5 / gfx1250, wave32. bf16 WMMA 16x16x32 with f32 accumulation.
typedef __attribute__((ext_vector_type(16))) __bf16 v16bf;
typedef __attribute__((ext_vector_type(8)))  __bf16 v8bf;
typedef __attribute__((ext_vector_type(8)))  float  v8f;

// ---------------------------------------------------------------------------
// Generic tiled GEMM:  O[row, n] = act( sum_k X[row,k] * W[n,k] + bias[n] (+extra) )
// Wave computes a 16(M) x 64(N) tile; block = 8 waves = 128 rows x 64 cols.
// A layout (16-bit A 16x32): lane m=(l&15); lanes 0-15 hold K 0..15,
// lanes 16-31 hold K 16..31  -> one contiguous 16-element run per lane.
// B layout (32x16, B[k][n]=W[n][k]): lane n holds W[n][k..] contiguous;
// lanes 0-15 -> K 0..15, lanes 16-31 -> K 16..31 (one 32B chunk per lane).
// ---------------------------------------------------------------------------
template <bool XF32>
__global__ __launch_bounds__(256) void wmma_gemm_k(
    const float* __restrict__ Xf, const __bf16* __restrict__ Xb, int ldx,
    const __bf16* __restrict__ W, int ldw, int K,
    const float* __restrict__ bias,
    const float* __restrict__ extra, int extra_mode,   // 1: += extra[(b*64+w)*ldo + col]
    float* __restrict__ outF, __bf16* __restrict__ outB, int ldo,
    int rows, int swap_vw, int relu_act)
{
  const int lane    = threadIdx.x & 31;
  const int wave    = threadIdx.x >> 5;
  const int rowtile = blockIdx.x * 128 + wave * 16;
  const int colbase = blockIdx.y * 64;
  if (rowtile >= rows) return;

  // A-source row for this lane (swap maps (b,v,w)->(b,w,v) for iter-2 transpose)
  int arow = rowtile + (lane & 15);
  if (swap_vw) {
    const int b = arow >> 12, v = (arow >> 6) & 63, wq = arow & 63;
    arow = (b << 12) | (wq << 6) | v;
  }
  const int ksel = lane >> 4;   // half-wave select
  const int aoff = ksel * 16;   // A chunk K offset
  const int boff = ksel * 16;   // B chunk K offset
  const int col_lane = colbase + (lane & 15);

  v8f acc[4] = {};

  for (int k0 = 0; k0 < K; k0 += 32) {
    v16bf a;
    if (XF32) {
      const float* pa = Xf + (size_t)arow * ldx + k0 + aoff;
      const float4 c0 = ((const float4*)pa)[0];
      const float4 c1 = ((const float4*)pa)[1];
      const float4 c2 = ((const float4*)(pa + 8))[0];
      const float4 c3 = ((const float4*)(pa + 8))[1];
      a[0]  = (__bf16)c0.x; a[1]  = (__bf16)c0.y; a[2]  = (__bf16)c0.z; a[3]  = (__bf16)c0.w;
      a[4]  = (__bf16)c1.x; a[5]  = (__bf16)c1.y; a[6]  = (__bf16)c1.z; a[7]  = (__bf16)c1.w;
      a[8]  = (__bf16)c2.x; a[9]  = (__bf16)c2.y; a[10] = (__bf16)c2.z; a[11] = (__bf16)c2.w;
      a[12] = (__bf16)c3.x; a[13] = (__bf16)c3.y; a[14] = (__bf16)c3.z; a[15] = (__bf16)c3.w;
    } else {
      const __bf16* pa = Xb + (size_t)arow * ldx + k0 + aoff;
      union { v16bf v; v8bf h[2]; } u;
      u.h[0] = *(const v8bf*)pa;
      u.h[1] = *(const v8bf*)(pa + 8);
      a = u.v;
    }
#pragma unroll
    for (int t = 0; t < 4; ++t) {
      const v16bf bmat =
          *(const v16bf*)(W + (size_t)(col_lane + t * 16) * ldw + k0 + boff);
      acc[t] = __builtin_amdgcn_wmma_f32_16x16x32_bf16(
          false, a, false, bmat, (short)0, acc[t], false, false);
    }
  }

  // Epilogue: D layout is VGPR j -> M = j + (half? 8:0), N = lane&15 per 16-col tile.
  const int rof = ksel * 8;
#pragma unroll
  for (int t = 0; t < 4; ++t) {
    const int ocol = col_lane + t * 16;
    const float bv = bias ? bias[ocol] : 0.0f;
#pragma unroll
    for (int j = 0; j < 8; ++j) {
      const int orow = rowtile + rof + j;
      float val = acc[t][j] + bv;
      if (extra_mode == 1) {
        const int b = orow >> 12;
        const int wq = orow & 63;
        val += extra[(size_t)(b * 64 + wq) * ldo + ocol];
      }
      if (relu_act) val = fmaxf(val, 0.0f);
      if (outB) outB[(size_t)orow * ldo + ocol] = (__bf16)val;
      if (outF) outF[(size_t)orow * ldo + ocol] = val;
    }
  }
}

// fp32 -> bf16 weight conversion
__global__ void cvt_kernel(const float* __restrict__ in, __bf16* __restrict__ out, int n) {
  const int i = blockIdx.x * blockDim.x + threadIdx.x;
  if (i < n) out[i] = (__bf16)in[i];
}

// link output layer: pred[row] = dot(h2[row, :512], Wlo) + blo   (wave per row)
__global__ __launch_bounds__(256) void pred_kernel(
    const __bf16* __restrict__ H, const float* __restrict__ Wlo,
    const float* __restrict__ blo, float* __restrict__ pred, int rows)
{
  const int row = blockIdx.x * 8 + (threadIdx.x >> 5);
  if (row >= rows) return;
  const int lane = threadIdx.x & 31;
  const __bf16* p = H + (size_t)row * 512 + lane * 16;
  const float* wq = Wlo + lane * 16;
  float acc = 0.0f;
#pragma unroll
  for (int i = 0; i < 16; ++i) acc += (float)p[i] * wq[i];
#pragma unroll
  for (int off = 16; off > 0; off >>= 1) acc += __shfl_xor(acc, off, 32);
  if (lane == 0) pred[row] = acc + blo[0];
}

// gated[row, m] = sigmoid(pred[row]) * M[row, m]
__global__ __launch_bounds__(256) void gate_kernel(
    const float* __restrict__ pred, const __bf16* __restrict__ M,
    __bf16* __restrict__ G, int total)
{
  const int i = blockIdx.x * blockDim.x + threadIdx.x;
  if (i >= total) return;
  const float s = 1.0f / (1.0f + __expf(-pred[i >> 9]));
  G[i] = (__bf16)(s * (float)M[i]);
}

// x[(b,v), m] = sum_w gated[(b,v,w), m]
__global__ __launch_bounds__(256) void msum_kernel(
    const __bf16* __restrict__ G, __bf16* __restrict__ X, int total)
{
  const int o = blockIdx.x * blockDim.x + threadIdx.x;
  if (o >= total) return;
  const int bv = o >> 9;
  const int m  = o & 511;
  const __bf16* p = G + (size_t)bv * 64 * 512 + m;
  float acc = 0.0f;
#pragma unroll 8
  for (int wq = 0; wq < 64; ++wq) acc += (float)p[wq * 512];
  X[o] = (__bf16)acc;
}

// GRU cell (torch gate order r,z,n) + readout FCs, one block per (b,v)
__global__ __launch_bounds__(256) void gru_readout_kernel(
    const float* __restrict__ gi, const float* __restrict__ gh,
    const float* __restrict__ nf,
    const float* __restrict__ Wr1, const float* __restrict__ br1,
    const float* __restrict__ Wr2, const float* __restrict__ br2,
    float* __restrict__ labels, float* __restrict__ roles)
{
  __shared__ float h[512];
  const int row = blockIdx.x;
  const int t = threadIdx.x;
  const float* gir = gi + (size_t)row * 1536;
  const float* ghr = gh + (size_t)row * 1536;
  for (int k = t; k < 512; k += 256) {
    const float r = 1.0f / (1.0f + __expf(-(gir[k] + ghr[k])));
    const float z = 1.0f / (1.0f + __expf(-(gir[512 + k] + ghr[512 + k])));
    const float n = tanhf(gir[1024 + k] + r * ghr[1024 + k]);
    const float hp = nf[(size_t)row * 512 + k];
    h[k] = (1.0f - z) * n + z * hp;
  }
  __syncthreads();
  if (t < 26) {
    float a = br1[t];
    const float* wr = Wr1 + t * 512;
    for (int k = 0; k < 512; ++k) a += h[k] * wr[k];
    labels[row * 26 + t] = a;
  } else if (t < 28) {
    const int o = t - 26;
    float a = br2[o];
    const float* wr = Wr2 + o * 512;
    for (int k = 0; k < 512; ++k) a += h[k] * wr[k];
    roles[row * 2 + o] = a;
  }
}

extern "C" void kernel_launch(void* const* d_in, const int* in_sizes, int n_in,
                              void* d_out, int out_size, void* d_ws, size_t ws_size,
                              hipStream_t stream)
{
  const float* edge = (const float*)d_in[0];   // [16,64,64,512]
  const float* nf   = (const float*)d_in[1];   // [16,64,512]
  const float* Wm   = (const float*)d_in[4];   // [512,1024]
  const float* bm   = (const float*)d_in[5];
  const float* Wl1  = (const float*)d_in[6];   // [512,512]
  const float* bl1  = (const float*)d_in[7];
  const float* Wl2  = (const float*)d_in[8];
  const float* bl2  = (const float*)d_in[9];
  const float* Wlo  = (const float*)d_in[10];  // [1,512]
  const float* blo  = (const float*)d_in[11];
  const float* Wih  = (const float*)d_in[12];  // [1536,512]
  const float* bih  = (const float*)d_in[13];
  const float* Whh  = (const float*)d_in[14];  // [1536,512]
  const float* bhh  = (const float*)d_in[15];
  const float* Wr1  = (const float*)d_in[16];  // [26,512]
  const float* br1  = (const float*)d_in[17];
  const float* Wr2  = (const float*)d_in[18];  // [2,512]
  const float* br2  = (const float*)d_in[19];

  float* out    = (float*)d_out;
  float* pred2  = out;                      // [16,64,64] = 65536
  float* labels = out + 65536;              // [16,64,26] = 26624
  float* roles  = out + 65536 + 26624;      // [16,64,2]  = 2048

  char* wsp = (char*)d_ws;
  auto alloc = [&](size_t bytes) -> void* {
    void* p = (void*)wsp;
    wsp += (bytes + 255) & ~(size_t)255;
    return p;
  };
  __bf16* WmB    = (__bf16*)alloc((size_t)512 * 1024 * 2);
  __bf16* Wl1B   = (__bf16*)alloc((size_t)512 * 512 * 2);
  __bf16* Wl2B   = (__bf16*)alloc((size_t)512 * 512 * 2);
  __bf16* WihB   = (__bf16*)alloc((size_t)1536 * 512 * 2);
  __bf16* WhhB   = (__bf16*)alloc((size_t)1536 * 512 * 2);
  float*  nfpart = (float*) alloc((size_t)1024 * 512 * 4);
  float*  pred1  = (float*) alloc((size_t)65536 * 4);
  __bf16* xB     = (__bf16*)alloc((size_t)1024 * 512 * 2);
  float*  gi     = (float*) alloc((size_t)1024 * 1536 * 4);
  float*  gh     = (float*) alloc((size_t)1024 * 1536 * 4);
  __bf16* Mb     = (__bf16*)alloc((size_t)65536 * 512 * 2);  // messages M
  __bf16* H1     = (__bf16*)alloc((size_t)65536 * 512 * 2);  // link hidden 1
  __bf16* SH     = (__bf16*)alloc((size_t)65536 * 512 * 2);  // h2 / gated (aliased)

  // weights -> bf16
  cvt_kernel<<<(512 * 1024 + 255) / 256, 256, 0, stream>>>(Wm,  WmB,  512 * 1024);
  cvt_kernel<<<(512 * 512 + 255) / 256, 256, 0, stream>>>(Wl1, Wl1B, 512 * 512);
  cvt_kernel<<<(512 * 512 + 255) / 256, 256, 0, stream>>>(Wl2, Wl2B, 512 * 512);
  cvt_kernel<<<(1536 * 512 + 255) / 256, 256, 0, stream>>>(Wih, WihB, 1536 * 512);
  cvt_kernel<<<(1536 * 512 + 255) / 256, 256, 0, stream>>>(Whh, WhhB, 1536 * 512);

  const dim3 blk(256);
  // node part of message: nfpart[(b,w), m] = Wm[:, :512] . nf[b,w,:]
  wmma_gemm_k<true><<<dim3(8, 8), blk, 0, stream>>>(
      nf, nullptr, 512, WmB, 1024, 512, nullptr, nullptr, 0,
      nfpart, nullptr, 512, 1024, 0, 0);
  // M[(b,v,w), m] = relu(Wm[:,512:] . e + nfpart[(b,w)] + bm)
  wmma_gemm_k<true><<<dim3(512, 8), blk, 0, stream>>>(
      edge, nullptr, 512, WmB + 512, 1024, 512, bm, nfpart, 1,
      nullptr, Mb, 512, 65536, 0, 1);

  // ---- propagation round 1 (edge_state = e) ----
  wmma_gemm_k<true><<<dim3(512, 8), blk, 0, stream>>>(
      edge, nullptr, 512, Wl1B, 512, 512, bl1, nullptr, 0,
      nullptr, H1, 512, 65536, 0, 1);
  wmma_gemm_k<false><<<dim3(512, 8), blk, 0, stream>>>(
      nullptr, H1, 512, Wl2B, 512, 512, bl2, nullptr, 0,
      nullptr, SH, 512, 65536, 0, 1);
  pred_kernel<<<8192, 256, 0, stream>>>(SH, Wlo, blo, pred1, 65536);
  gate_kernel<<<(65536 * 512) / 256, 256, 0, stream>>>(pred1, Mb, SH, 65536 * 512);

  // ---- propagation round 2 (edge_state = gated^T via swap_vw) ----
  wmma_gemm_k<false><<<dim3(512, 8), blk, 0, stream>>>(
      nullptr, SH, 512, Wl1B, 512, 512, bl1, nullptr, 0,
      nullptr, H1, 512, 65536, 1, 1);
  wmma_gemm_k<false><<<dim3(512, 8), blk, 0, stream>>>(
      nullptr, H1, 512, Wl2B, 512, 512, bl2, nullptr, 0,
      nullptr, SH, 512, 65536, 0, 1);
  pred_kernel<<<8192, 256, 0, stream>>>(SH, Wlo, blo, pred2, 65536);   // output 1
  gate_kernel<<<(65536 * 512) / 256, 256, 0, stream>>>(pred2, Mb, SH, 65536 * 512);

  // m_sum over w -> GRU input x
  msum_kernel<<<(1024 * 512) / 256, 256, 0, stream>>>(SH, xB, 1024 * 512);

  // GRU gate GEMMs
  wmma_gemm_k<false><<<dim3(8, 24), blk, 0, stream>>>(
      nullptr, xB, 512, WihB, 512, 512, bih, nullptr, 0,
      gi, nullptr, 1536, 1024, 0, 0);
  wmma_gemm_k<true><<<dim3(8, 24), blk, 0, stream>>>(
      nf, nullptr, 512, WhhB, 512, 512, bhh, nullptr, 0,
      gh, nullptr, 1536, 1024, 0, 0);

  // GRU cell + readouts (outputs 2 and 3)
  gru_readout_kernel<<<1024, 256, 0, stream>>>(gi, gh, nf, Wr1, br1, Wr2, br2,
                                               labels, roles);
}
